// RNN_19439021981759
// MI455X (gfx1250) — compile-verified, matrix operand order checked by reading, not run on hardware
//
#include <hip/hip_runtime.h>
#include <hip/hip_bf16.h>
#include <math.h>

// ---------------------------------------------------------------------------
// Persistent-kernel bf16-WMMA RNN for MI455X (gfx1250, wave32).
//   h_t = tanh(embed[x_t] + h_{t-1} @ W_hh + b_h)
// N=32 batch, H=1024 hidden, T=2048 steps.
//
// - W_hh resident in LDS (bf16, transposed, 16B-chunk xor-swizzled),
//   64KB slice per WG, loaded once.
// - h state ping-pongs through L2, grid-barriered per step.
// - Two-stage lookahead: tokens for t+2 and embed rows for t+1 are in
//   flight while step t computes -> zero gather latency on critical path.
// - 128 waves = 32 WGs x 4; one 16x16 output tile per wave;
//   K-loop: 32 x v_wmma_f32_16x16x32_bf16.
// ---------------------------------------------------------------------------

typedef __attribute__((ext_vector_type(16))) __bf16 v16bf;
typedef __attribute__((ext_vector_type(8)))  __bf16 v8bf;
typedef __attribute__((ext_vector_type(8)))  float  v8f;

#define HIDDEN      1024
#define NBATCH      32
#define SEQT        2048
#define NWG         32
#define NTHREADS    128            // 4 wave32 per WG
#define COLS_PER_WG 32             // HIDDEN / NWG -> 64KB bf16 LDS slice

__device__ __forceinline__ float fast_tanh(float v) {
#if __has_builtin(__builtin_amdgcn_tanhf)
    return __builtin_amdgcn_tanhf(v);          // native v_tanh_f32
#elif __has_builtin(__builtin_amdgcn_tanh_f32)
    return __builtin_amdgcn_tanh_f32(v);
#else
    return tanhf(v);                           // ocml fallback
#endif
}

__device__ __forceinline__ void grid_barrier(unsigned* ctr, unsigned target) {
    __threadfence();
    __syncthreads();
    if (threadIdx.x == 0) {
        __hip_atomic_fetch_add(ctr, 1u, __ATOMIC_ACQ_REL, __HIP_MEMORY_SCOPE_AGENT);
        while (__hip_atomic_load(ctr, __ATOMIC_ACQUIRE, __HIP_MEMORY_SCOPE_AGENT) < target) {
            __builtin_amdgcn_s_sleep(2);
        }
    }
    __syncthreads();
}

__global__ __launch_bounds__(NTHREADS)
void rnn_wmma_persistent_19439021981759(const int*   __restrict__ x,      // [N,T]
                                        const float* __restrict__ embed,  // [V,H]
                                        const float* __restrict__ W,      // [H,H] row-major (k, n)
                                        const float* __restrict__ bh,     // [H]
                                        float*       __restrict__ z,      // [N,T,H]
                                        float*       __restrict__ hlast,  // [N,H]
                                        __bf16*      __restrict__ hbuf0,  // [N,H] bf16 state ping
                                        __bf16*      __restrict__ hbuf1,  // [N,H] bf16 state pong
                                        unsigned*    __restrict__ barrier_ctr)
{
    // W^T slice for this WG's 32 output columns: Wt[c][k], bf16, 64KB.
    // 16B chunks xor-swizzled by (c & 15) to kill ds bank conflicts.
    __shared__ __bf16 Wt[COLS_PER_WG * HIDDEN];

    const int wg   = blockIdx.x;
    const int tid  = threadIdx.x;
    const int col0 = wg * COLS_PER_WG;

    // ---- one-time: stage W slice into LDS (transpose + bf16 + swizzle) ----
    for (int idx = tid; idx < COLS_PER_WG * HIDDEN; idx += NTHREADS) {
        int c = idx & (COLS_PER_WG - 1);          // consecutive tids -> consecutive cols
        int k = idx >> 5;                         // -> coalesced global row reads
        float w = W[(size_t)k * HIDDEN + (col0 + c)];
        int chunk = k >> 3;                       // 16B chunk index along K
        int kk = ((chunk ^ (c & 15)) << 3) | (k & 7);
        Wt[c * HIDDEN + kk] = (__bf16)w;
    }
    // ---- one-time: h0 = 0 for this WG's columns ----
    for (int i = tid; i < NBATCH * COLS_PER_WG; i += NTHREADS) {
        int m = i >> 5;
        int c = i & (COLS_PER_WG - 1);
        hbuf0[(size_t)m * HIDDEN + col0 + c] = (__bf16)0.0f;
    }

    const int lane   = tid & 31;
    const int wave   = tid >> 5;
    const int laneLo = lane & 15;
    const int laneHi = lane >> 4;
    const int mtile  = wave & 1;                  // which 16 batch rows
    const int ctile  = wave >> 1;                 // which 16 of our 32 columns
    const int clocal = ctile * 16 + laneLo;       // local column (B/C lane column)
    const int colg   = col0 + clocal;             // global hidden column
    const int arow   = mtile * 16 + laneLo;       // batch row feeding A fragment
    const int mbase  = mtile * 16 + laneHi * 8;   // first batch row this lane covers in C

    const float bias = bh[colg];
    const int   csw  = clocal & 15;               // swizzle key

    // ---- two-stage lookahead pipeline (independent of h):
    //      nev[]  = embed values for step t   (in flight across barrier)
    //      ntok[] = tokens for step t+1       (in flight across barrier)
    float ev[8], nev[8];
    int   ntok[8];
#pragma unroll
    for (int j = 0; j < 8; ++j) {
        int tok0 = x[(size_t)(mbase + j) * SEQT + 0];
        nev[j]  = embed[(size_t)tok0 * HIDDEN + colg];
    }
#pragma unroll
    for (int j = 0; j < 8; ++j) {
        ntok[j] = x[(size_t)(mbase + j) * SEQT + 1];
    }

    unsigned epoch = 1;
    grid_barrier(barrier_ctr, epoch * NWG);       // staging + h0 ready

    for (int t = 0; t < SEQT; ++t) {
        // consume embed values prefetched during the previous iteration
        // (single batched wait, hidden behind the barrier spin)
#pragma unroll
        for (int j = 0; j < 8; ++j) ev[j] = nev[j];

        const __bf16* hprev = (t & 1) ? hbuf1 : hbuf0;
        __bf16*       hnext = (t & 1) ? hbuf0 : hbuf1;

        v8f acc = {};
        const __bf16* arow_p = hprev + (size_t)arow * HIDDEN + laneHi * 8;
        const int bbase = clocal * HIDDEN;

#pragma unroll 4
        for (int k0 = 0; k0 < HIDDEN; k0 += 32) {
            union { v16bf v; struct { v8bf lo, hi; } s; } a, b;
            // A: K = k0 + 8*laneHi + {0..7}  and  + 16 + {0..7}
            a.s.lo = *(const v8bf*)(arow_p + k0);
            a.s.hi = *(const v8bf*)(arow_p + k0 + 16);
            // B: column clocal of W, same K pattern, de-swizzled chunk lookup
            int chunk = (k0 >> 3) + laneHi;
            b.s.lo = *(const v8bf*)&Wt[bbase + (((chunk    ) ^ csw) << 3)];
            b.s.hi = *(const v8bf*)&Wt[bbase + (((chunk + 2) ^ csw) << 3)];
            acc = __builtin_amdgcn_wmma_f32_16x16x32_bf16(
                      false, a.v, false, b.v, (short)0, acc, false, false);
        }

        // ---- issue next step's embedding gather from already-resident
        //      tokens (no wait), then refill the token stage for t+2 ----
        if (t + 1 < SEQT) {
#pragma unroll
            for (int j = 0; j < 8; ++j) {
                nev[j] = embed[(size_t)ntok[j] * HIDDEN + colg];
            }
            if (t + 2 < SEQT) {
#pragma unroll
                for (int j = 0; j < 8; ++j) {
                    ntok[j] = x[(size_t)(mbase + j) * SEQT + (t + 2)];
                }
            }
        }

        // C layout: element j of lane -> (M = mbase + j, N = laneLo)
#pragma unroll
        for (int j = 0; j < 8; ++j) {
            int m   = mbase + j;                               // batch index
            float v = acc[j] + ev[j] + bias;
            float h = fast_tanh(v);
            z[((size_t)m * SEQT + t) * HIDDEN + colg] = h;
            hnext[(size_t)m * HIDDEN + colg] = (__bf16)h;
            if (t == SEQT - 1) {
                hlast[(size_t)m * HIDDEN + colg] = h;
            }
        }

        ++epoch;
        grid_barrier(barrier_ctr, epoch * NWG);   // h_next globally visible
    }
}

extern "C" void kernel_launch(void* const* d_in, const int* in_sizes, int n_in,
                              void* d_out, int out_size, void* d_ws, size_t ws_size,
                              hipStream_t stream) {
    (void)in_sizes; (void)n_in; (void)out_size; (void)ws_size;

    const int*   x     = (const int*)  d_in[0];   // [32, 2048] int32
    const float* embed = (const float*)d_in[1];   // [50257, 1024] f32
    const float* W     = (const float*)d_in[2];   // [1024, 1024] f32
    const float* bh    = (const float*)d_in[3];   // [1024] f32

    float* z     = (float*)d_out;                               // [32,2048,1024]
    float* hlast = z + (size_t)NBATCH * SEQT * HIDDEN;          // [32,1024]

    unsigned* ctr  = (unsigned*)d_ws;
    __bf16*   hb0  = (__bf16*)((char*)d_ws + 256);
    __bf16*   hb1  = (__bf16*)((char*)d_ws + 256 + (size_t)NBATCH * HIDDEN * sizeof(__bf16));

    // reset grid barrier counter every launch (graph-capturable)
    hipMemsetAsync(d_ws, 0, 256, stream);

    rnn_wmma_persistent_19439021981759<<<NWG, NTHREADS, 0, stream>>>(
        x, embed, W, bh, z, hlast, hb0, hb1, ctr);
}
